// PNbatch_62405874811853
// MI455X (gfx1250) — compile-verified, matrix operand order checked by reading, not run on hardware
//
#include <hip/hip_runtime.h>

// ---------------- CDNA5 WMMA types ----------------
typedef __attribute__((ext_vector_type(16))) __bf16 v16bf;
typedef __attribute__((ext_vector_type(8)))  float  v8f;

union Frag16 {            // 32 bytes: two 16B halves -> one 16-element bf16 fragment
    struct { uint4 lo, hi; } u;
    v16bf v;
};

__device__ __forceinline__ unsigned short f32_to_bf16(float f) {
    unsigned int u = __float_as_uint(f);
    u += 0x7FFFu + ((u >> 16) & 1u);      // round-to-nearest-even
    return (unsigned short)(u >> 16);
}

// ---------------- problem constants ----------------
#define NPTS   131072
#define NSEG   1024
#define EMB    1024
#define MTILE  64            // rows (points) per workgroup: 4x B-fragment reuse
#define RSTRIDE 1024         // LDS act row stride in bf16 elems (bank-safe via XOR swizzle)
#define REDSTRIDE 260        // f32 reduction row stride (bank spread)

// per-layer metadata (fin padded to mult. of 32 for uniform 16x16x32 WMMA)
#define L_FIN  {3, 64, 64, 64, 128, 1024, 512, 256}
#define L_FINP {32, 64, 64, 64, 128, 1024, 512, 256}
#define L_FOUT {64, 64, 64, 128, 1024, 512, 256, 1024}
#define L_WOFF {0, 2048, 6144, 10240, 18432, 149504, 673792, 804864}
#define L_BOFF {0, 64, 128, 192, 320, 1344, 1856, 2112}
#define W_TOTAL  1067008     // bf16 elements
#define B_TOTAL  3136        // f32 elements

// XOR swizzle: 16B chunks rotated per row so 16 rows hit 16 distinct bank groups
__device__ __forceinline__ int swz(int r, int c) {
    return r * RSTRIDE + ((((c >> 3) ^ (r & 15)) << 3) | (c & 7));
}

// ---------------- prep kernels ----------------
__global__ void init_emb_kernel(float* emb) {
    int i = blockIdx.x * blockDim.x + threadIdx.x;
    emb[i] = -3.402823466e38f;   // -FLT_MAX (segment max identity)
}

// W [fin][fout] f32 -> Wt [fout][fin_pad] bf16 (zero padded K)
__global__ void cvt_w_kernel(const float* __restrict__ W, unsigned short* __restrict__ dst,
                             int fin, int fin_pad, int fout) {
    int idx = blockIdx.x * blockDim.x + threadIdx.x;
    if (idx >= fout * fin_pad) return;
    int n = idx / fin_pad;
    int k = idx - n * fin_pad;
    unsigned short r = 0;
    if (k < fin) r = f32_to_bf16(W[k * fout + n]);
    dst[idx] = r;
}

__global__ void cvt_b_kernel(const float* __restrict__ b, float* __restrict__ dst, int n) {
    int i = blockIdx.x * blockDim.x + threadIdx.x;
    if (i < n) dst[i] = b[i];
}

// ---------------- fused MLP + segmented max ----------------
// 1 workgroup = 16 wave32 = one 64-point tile. Activations ping inside one
// 128KB dynamic-LDS buffer (CDNA5 allows up to 320KB/workgroup). Each B
// fragment streamed from L2-resident weights feeds 4 WMMAs (4 row sub-tiles).
__global__ __launch_bounds__(512) void mlp_fused_kernel(
    const float* __restrict__ x, const int* __restrict__ seg,
    const unsigned short* __restrict__ wbf, const float* __restrict__ bias,
    float* __restrict__ emb) {

    extern __shared__ char smem[];                     // 128 KB dynamic
    unsigned short* act = (unsigned short*)smem;       // [64][1024] bf16 (swizzled)
    float* red = (float*)smem;                         // reused: [64][REDSTRIDE] f32
    __shared__ int segtile[MTILE];

    const int tid  = threadIdx.x;
    const int lane = tid & 31;
    const int wave = tid >> 5;                         // 0..15
    const int nrow = lane & 15;                        // A: row / B,D: col
    const int hi   = lane >> 4;                        // half-wave K/M split
    const int base = blockIdx.x * MTILE;

    if (tid < MTILE) segtile[tid] = seg[base + tid];

    // ---- stage x into LDS, zero-padded K=3 -> 32, swizzled ----
    for (int i = tid; i < MTILE * 32; i += 512) {
        int r = i >> 5, c = i & 31;
        unsigned short v = 0;
        if (c < 3) v = f32_to_bf16(x[(base + r) * 3 + c]);
        act[swz(r, c)] = v;
    }
    __syncthreads();

    const int FINP[8] = L_FINP;
    const int FOUT[8] = L_FOUT;
    const int WOFF[8] = L_WOFF;
    const int BOFF[8] = L_BOFF;

#pragma unroll
    for (int l = 0; l < 8; ++l) {
        const int finp   = FINP[l];
        const int fout   = FOUT[l];
        const int ntiles = fout >> 4;
        const unsigned short* wl = wbf + WOFF[l];
        const float* bl = bias + BOFF[l];

        v8f acc[4][4];                                 // [row sub-tile][col tile]
        {
            v8f z = {};
#pragma unroll
            for (int s = 0; s < 4; ++s)
#pragma unroll
                for (int j = 0; j < 4; ++j) acc[s][j] = z;
        }

        // ---- compute: k-chunks of 32; A loaded once, B reused by 4 sub-tiles
        for (int k0 = 0; k0 < finp; k0 += 32) {
            Frag16 A[4];
            const int c0 = (k0 >> 3) + hi;
#pragma unroll
            for (int s = 0; s < 4; ++s) {
                const int r = s * 16 + nrow;
                A[s].u.lo = *(const uint4*)(act + r * RSTRIDE + (((c0    ) ^ nrow) << 3));
                A[s].u.hi = *(const uint4*)(act + r * RSTRIDE + (((c0 + 2) ^ nrow) << 3));
            }
#pragma unroll
            for (int j = 0; j < 4; ++j) {
                const int t = wave + 16 * j;
                if (t < ntiles) {
                    Frag16 B;
                    const unsigned short* bp = wl + (t * 16 + nrow) * finp + k0 + hi * 8;
                    B.u.lo = *(const uint4*)(bp);
                    B.u.hi = *(const uint4*)(bp + 16);
#pragma unroll
                    for (int s = 0; s < 4; ++s)
                        acc[s][j] = __builtin_amdgcn_wmma_f32_16x16x32_bf16(
                                        false, A[s].v, false, B.v, (short)0,
                                        acc[s][j], false, false);
                }
            }
        }
        __syncthreads();   // all reads of act complete

        if (l < 7) {
            // ---- bias + ReLU + bf16 pack back into LDS (swizzled) ----
#pragma unroll
            for (int j = 0; j < 4; ++j) {
                const int t = wave + 16 * j;
                if (t < ntiles) {
                    const int col = t * 16 + nrow;
                    const float bv = bl[col];
#pragma unroll
                    for (int s = 0; s < 4; ++s)
#pragma unroll
                        for (int r = 0; r < 8; ++r) {
                            const float v = fmaxf(acc[s][j][r] + bv, 0.0f);
                            const int m = s * 16 + hi * 8 + r;       // D row layout
                            act[swz(m, col)] = f32_to_bf16(v);
                        }
                }
            }
        } else {
            // ---- final layer: LDS pre-reduction, then few atomics per segment.
            // seg_ids sorted, min run length 65 -> <=2 runs per 64-row tile,
            // cutting global_atomic_max_num_f32 count by ~25x.
            for (int g = 0; g < 4; ++g) {              // 4 column groups of 256
                const int t    = wave + 16 * g;        // exactly 1 tile/wave/group
                const int gcol = t * 16 + nrow;
                const int lcol = wave * 16 + nrow;     // 0..255
                const float bv = bl[gcol];
#pragma unroll
                for (int s = 0; s < 4; ++s)
#pragma unroll
                    for (int r = 0; r < 8; ++r) {
                        const int m = s * 16 + hi * 8 + r;
                        red[m * REDSTRIDE + lcol] = acc[s][g][r] + bv;  // no ReLU
                    }
                __syncthreads();
                if (tid < 256) {
                    const int col = tid + 256 * g;
                    int   cur  = segtile[0];
                    float runm = red[tid];
                    for (int m = 1; m < MTILE; ++m) {
                        const int   sgm = segtile[m];
                        const float v   = red[m * REDSTRIDE + tid];
                        if (sgm == cur) {
                            runm = fmaxf(runm, v);
                        } else {
                            atomicMax(&emb[cur * EMB + col], runm);
                            cur  = sgm;
                            runm = v;
                        }
                    }
                    atomicMax(&emb[cur * EMB + col], runm);
                }
                __syncthreads();
            }
        }
        __syncthreads();
    }
}

// ---------------- final FC: out = relu(emb) @ Wfc + bfc ----------------
__global__ void fc_kernel(const float* __restrict__ emb, const float* __restrict__ Wfc,
                          const float* __restrict__ bfc, float* __restrict__ out) {
    __shared__ float red[256];
    const int row = blockIdx.x;
    const int tid = threadIdx.x;           // 128 threads
    float s0 = 0.0f, s1 = 0.0f;
    for (int k = tid; k < EMB; k += 128) {
        float e = fmaxf(emb[row * EMB + k], 0.0f);
        s0 += e * Wfc[k * 2 + 0];
        s1 += e * Wfc[k * 2 + 1];
    }
    red[tid] = s0;
    red[128 + tid] = s1;
    __syncthreads();
    for (int off = 64; off > 0; off >>= 1) {
        if (tid < off) {
            red[tid]       += red[tid + off];
            red[128 + tid] += red[128 + tid + off];
        }
        __syncthreads();
    }
    if (tid == 0) {
        out[row * 2 + 0] = red[0]   + bfc[0];
        out[row * 2 + 1] = red[128] + bfc[1];
    }
}

// ---------------- launch ----------------
extern "C" void kernel_launch(void* const* d_in, const int* in_sizes, int n_in,
                              void* d_out, int out_size, void* d_ws, size_t ws_size,
                              hipStream_t stream) {
    const float* x   = (const float*)d_in[0];
    const int*   seg = (const int*)d_in[1];
    // d_in[2] = lengths, d_in[3] = edge_index  (unused by the reference math)
    const float* W[8];
    const float* B[8];
    for (int i = 0; i < 8; ++i) {
        W[i] = (const float*)d_in[4 + 2 * i];
        B[i] = (const float*)d_in[5 + 2 * i];
    }
    const float* Wfc = (const float*)d_in[20];
    const float* bfc = (const float*)d_in[21];

    // workspace layout
    float*          emb  = (float*)d_ws;                                      // 4 MB
    unsigned short* wbf  = (unsigned short*)((char*)d_ws + (size_t)4 * 1024 * 1024);
    float*          bias = (float*)((char*)d_ws + (size_t)4 * 1024 * 1024
                                                + (size_t)W_TOTAL * 2);       // 16B aligned

    init_emb_kernel<<<(NSEG * EMB) / 256, 256, 0, stream>>>(emb);

    const int FIN[8]  = L_FIN;
    const int FINP[8] = L_FINP;
    const int FOUT[8] = L_FOUT;
    const int WOFF[8] = L_WOFF;
    const int BOFF[8] = L_BOFF;
    for (int l = 0; l < 8; ++l) {
        const int tot = FOUT[l] * FINP[l];
        cvt_w_kernel<<<(tot + 255) / 256, 256, 0, stream>>>(
            W[l], wbf + WOFF[l], FIN[l], FINP[l], FOUT[l]);
        cvt_b_kernel<<<(FOUT[l] + 255) / 256, 256, 0, stream>>>(
            B[l], bias + BOFF[l], FOUT[l]);
    }

    // 128KB dynamic LDS per workgroup (CDNA5 WGP has 320KB)
    mlp_fused_kernel<<<NPTS / MTILE, 512, 131072, stream>>>(x, seg, wbf, bias, emb);
    fc_kernel<<<NSEG, 128, 0, stream>>>(emb, Wfc, bfc, (float*)d_out);
}
